// LinOSSBlock_54614804136684
// MI455X (gfx1250) — matrix operand order repaired and checked by reading, not compiled, not run on hardware
//
#include <hip/hip_runtime.h>
#include <hip/hip_bf16.h>
#include <math.h>

// Problem constants (match reference)
#define MD 128                    // hidden dim
#define NT 8192                   // timesteps
#define BB 16                     // batch
#define RTOT (BB * NT)            // 131072 GEMM rows
#define CCH 128                   // scan chunks
#define LCH (NT / CCH)            // 64 steps per chunk

typedef __attribute__((ext_vector_type(16))) _Float16 v16h;
typedef __attribute__((ext_vector_type(8)))  float    v8f;

__device__ __forceinline__ v8f wmma_f16(v16h a, v16h b, v8f c) {
  // D(16x16 f32) = A(16x32 f16) * B(32x16 f16) + C
  return __builtin_amdgcn_wmma_f32_16x16x32_f16(false, a, false, b, (short)0, c,
                                                false, false);
}

// ---- fragment builders (ISA 7.12.2 layouts, wave32) -------------------------

// A-matrix 16x32 f16 from fp32 row-major src: lane holds row (lane&15),
// K = kc*32 + {kg..kg+7, 16+kg..16+kg+7}, kg = (lane>>4)*8
__device__ __forceinline__ v16h a_frag_f32(const float* __restrict__ base,
                                           int row0, int kc) {
  int lane = threadIdx.x & 31;
  const float* p = base + (size_t)(row0 + (lane & 15)) * MD + kc * 32 + ((lane >> 4) << 3);
  v16h f;
#pragma unroll
  for (int i = 0; i < 8; ++i) { f[i] = (_Float16)p[i]; f[8 + i] = (_Float16)p[16 + i]; }
  return f;
}

__device__ __forceinline__ v16h a_frag_f16(const _Float16* __restrict__ base,
                                           int row0, int kc) {
  int lane = threadIdx.x & 31;
  const _Float16* p = base + (size_t)(row0 + (lane & 15)) * MD + kc * 32 + ((lane >> 4) << 3);
  v16h f;
#pragma unroll
  for (int i = 0; i < 8; ++i) { f[i] = p[i]; f[8 + i] = p[16 + i]; }
  return f;
}

// B-matrix 32x16 f16 from LDS weights W[n_out][k_in] row-major (out = act @ W^T):
// lane holds col n = nt*16 + (lane&15), K = kc*32 + {kg..kg+7, 16+kg..16+kg+7}, kg = (lane>>4)*8
__device__ __forceinline__ v16h b_frag_lds(const _Float16* w, int nt, int kc) {
  int lane = threadIdx.x & 31;
  const _Float16* p = w + (nt * 16 + (lane & 15)) * MD + kc * 32 + ((lane >> 4) << 3);
  v16h f;
#pragma unroll
  for (int i = 0; i < 8; ++i) { f[i] = p[i]; f[8 + i] = p[16 + i]; }
  return f;
}

__device__ __forceinline__ void stage_w(_Float16* dst, const float* __restrict__ src) {
  for (int i = threadIdx.x; i < MD * MD; i += blockDim.x) dst[i] = (_Float16)src[i];
}

// ---- K1: fz = u @ Bw^T + Bb  (DT == 1.0) -----------------------------------
__global__ void k_gemm_fz(const float* __restrict__ u, const float* __restrict__ Bw,
                          const float* __restrict__ Bb, float* __restrict__ fz) {
  extern __shared__ _Float16 smem[];
  stage_w(smem, Bw);
  __syncthreads();
  int wave = threadIdx.x >> 5, lane = threadIdx.x & 31;
  int row0 = blockIdx.x * 128 + wave * 16;
  v16h a[4];
#pragma unroll
  for (int kc = 0; kc < 4; ++kc) a[kc] = a_frag_f32(u, row0, kc);
#pragma unroll
  for (int nt = 0; nt < 8; ++nt) {
    v8f acc = {};
#pragma unroll
    for (int kc = 0; kc < 4; ++kc) acc = wmma_f16(a[kc], b_frag_lds(smem, nt, kc), acc);
    int col = nt * 16 + (lane & 15);
    float bias = Bb[col];
    int rbase = row0 + ((lane >> 4) << 3);
#pragma unroll
    for (int r = 0; r < 8; ++r)
      fz[(size_t)(rbase + r) * MD + col] = acc[r] + bias;
  }
}

// ---- scan helpers ----------------------------------------------------------
__device__ __forceinline__ void scan_coef(const float* __restrict__ A_diag, int m,
                                          float& czz, float& czy, float& cyz, float& cyy) {
  float a = fmaxf(A_diag[m], 0.0f);
  float s = 1.0f / (1.0f + a);       // dt = 1
  czz = s; czy = -s * a; cyz = s; cyy = s;
}

// ---- K2: per-chunk zero-init scan, record final (z,y) ----------------------
__global__ void k_scan_local(const float* __restrict__ fz,
                             const float* __restrict__ A_diag,
                             float2* __restrict__ S) {
  int tid = blockIdx.x * blockDim.x + threadIdx.x;      // BB*CCH*MD threads
  int m = tid & (MD - 1);
  int c = (tid >> 7) & (CCH - 1);
  int b = tid >> 14;
  float czz, czy, cyz, cyy;
  scan_coef(A_diag, m, czz, czy, cyz, cyy);
  const float* p = fz + ((size_t)b * NT + (size_t)c * LCH) * MD + m;
  float z = 0.f, y = 0.f;
#pragma unroll 4
  for (int t = 0; t < LCH; ++t) {
    float f = p[(size_t)t * MD];
    float zp = z + f;
    float zn = czz * zp + czy * y;
    y = cyz * zp + cyy * y;
    z = zn;
  }
  S[((size_t)b * CCH + c) * MD + m] = make_float2(z, y);
}

// ---- K3: combine chunk states sequentially (per b,m) -----------------------
__global__ void k_scan_prefix(const float* __restrict__ A_diag,
                              const float2* __restrict__ S,
                              float2* __restrict__ Vin) {
  int tid = blockIdx.x * blockDim.x + threadIdx.x;      // BB*MD threads
  int m = tid & (MD - 1);
  int b = tid >> 7;
  float czz, czy, cyz, cyy;
  scan_coef(A_diag, m, czz, czy, cyz, cyy);
  // P = M^LCH via repeated squaring (LCH = 64 -> 6 squarings)
  float p00 = czz, p01 = czy, p10 = cyz, p11 = cyy;
#pragma unroll
  for (int i = 0; i < 6; ++i) {
    float q00 = p00 * p00 + p01 * p10;
    float q01 = p00 * p01 + p01 * p11;
    float q10 = p10 * p00 + p11 * p10;
    float q11 = p10 * p01 + p11 * p11;
    p00 = q00; p01 = q01; p10 = q10; p11 = q11;
  }
  float2 v = make_float2(0.f, 0.f);
  for (int c = 0; c < CCH; ++c) {
    size_t idx = ((size_t)b * CCH + c) * MD + m;
    Vin[idx] = v;
    float2 sc = S[idx];
    float2 nv;
    nv.x = p00 * v.x + p01 * v.y + sc.x;
    nv.y = p10 * v.x + p11 * v.y + sc.y;
    v = nv;
  }
}

// ---- K4: replay chunks with true incoming state, emit x (f16) --------------
__global__ void k_scan_replay(const float* __restrict__ fz,
                              const float* __restrict__ A_diag,
                              const float2* __restrict__ Vin,
                              _Float16* __restrict__ x16) {
  int tid = blockIdx.x * blockDim.x + threadIdx.x;
  int m = tid & (MD - 1);
  int c = (tid >> 7) & (CCH - 1);
  int b = tid >> 14;
  float czz, czy, cyz, cyy;
  scan_coef(A_diag, m, czz, czy, cyz, cyy);
  float2 v = Vin[((size_t)b * CCH + c) * MD + m];
  float z = v.x, y = v.y;
  const float* p = fz + ((size_t)b * NT + (size_t)c * LCH) * MD + m;
  _Float16* q = x16 + ((size_t)b * NT + (size_t)c * LCH) * MD + m;
#pragma unroll 4
  for (int t = 0; t < LCH; ++t) {
    float f = p[(size_t)t * MD];
    float zp = z + f;
    float zn = czz * zp + czy * y;
    y = cyz * zp + cyy * y;
    z = zn;
    q[(size_t)t * MD] = (_Float16)y;
  }
}

// ---- K5: h = gelu(x @ Cw^T + Cb + u @ Dw^T + Db), f16 out ------------------
__global__ void k_gemm_h(const _Float16* __restrict__ x16, const float* __restrict__ u,
                         const float* __restrict__ Cw, const float* __restrict__ Cb,
                         const float* __restrict__ Dw, const float* __restrict__ Db,
                         _Float16* __restrict__ h16) {
  extern __shared__ _Float16 smem[];
  stage_w(smem, Cw);
  stage_w(smem + MD * MD, Dw);
  __syncthreads();
  int wave = threadIdx.x >> 5, lane = threadIdx.x & 31;
  int row0 = blockIdx.x * 128 + wave * 16;
  v16h ax[4], au[4];
#pragma unroll
  for (int kc = 0; kc < 4; ++kc) {
    ax[kc] = a_frag_f16(x16, row0, kc);
    au[kc] = a_frag_f32(u, row0, kc);
  }
#pragma unroll
  for (int nt = 0; nt < 8; ++nt) {
    v8f acc = {};
#pragma unroll
    for (int kc = 0; kc < 4; ++kc) acc = wmma_f16(ax[kc], b_frag_lds(smem, nt, kc), acc);
#pragma unroll
    for (int kc = 0; kc < 4; ++kc) acc = wmma_f16(au[kc], b_frag_lds(smem + MD * MD, nt, kc), acc);
    int col = nt * 16 + (lane & 15);
    float bias = Cb[col] + Db[col];
    int rbase = row0 + ((lane >> 4) << 3);
#pragma unroll
    for (int r = 0; r < 8; ++r) {
      float v = acc[r] + bias;
      float g = 0.5f * v * (1.0f + erff(v * 0.70710678118654752f)); // exact GELU
      h16[(size_t)(rbase + r) * MD + col] = (_Float16)g;
    }
  }
}

// ---- K6: out = (h @ W1^T + b1) * sigmoid(h @ W2^T + b2) + u ----------------
__global__ void k_gemm_out(const _Float16* __restrict__ h16, const float* __restrict__ u,
                           const float* __restrict__ W1, const float* __restrict__ b1,
                           const float* __restrict__ W2, const float* __restrict__ b2,
                           float* __restrict__ out) {
  extern __shared__ _Float16 smem[];
  stage_w(smem, W1);
  stage_w(smem + MD * MD, W2);
  __syncthreads();
  int wave = threadIdx.x >> 5, lane = threadIdx.x & 31;
  int row0 = blockIdx.x * 128 + wave * 16;
  v16h ah[4];
#pragma unroll
  for (int kc = 0; kc < 4; ++kc) ah[kc] = a_frag_f16(h16, row0, kc);
#pragma unroll
  for (int nt = 0; nt < 8; ++nt) {
    v8f acc1 = {}, acc2 = {};
#pragma unroll
    for (int kc = 0; kc < 4; ++kc) acc1 = wmma_f16(ah[kc], b_frag_lds(smem, nt, kc), acc1);
#pragma unroll
    for (int kc = 0; kc < 4; ++kc) acc2 = wmma_f16(ah[kc], b_frag_lds(smem + MD * MD, nt, kc), acc2);
    int col = nt * 16 + (lane & 15);
    float bb1 = b1[col], bb2 = b2[col];
    int rbase = row0 + ((lane >> 4) << 3);
#pragma unroll
    for (int r = 0; r < 8; ++r) {
      size_t idx = (size_t)(rbase + r) * MD + col;
      float g1 = acc1[r] + bb1;
      float g2 = acc2[r] + bb2;
      float sg = 1.0f / (1.0f + expf(-g2));
      out[idx] = g1 * sg + u[idx];
    }
  }
}

// ---- host side -------------------------------------------------------------
extern "C" void kernel_launch(void* const* d_in, const int* in_sizes, int n_in,
                              void* d_out, int out_size, void* d_ws, size_t ws_size,
                              hipStream_t stream) {
  const float* u      = (const float*)d_in[0];
  const float* A_diag = (const float*)d_in[1];
  const float* B_w    = (const float*)d_in[2];
  const float* B_b    = (const float*)d_in[3];
  const float* C_w    = (const float*)d_in[4];
  const float* C_b    = (const float*)d_in[5];
  const float* D_w    = (const float*)d_in[6];
  const float* D_b    = (const float*)d_in[7];
  const float* W1     = (const float*)d_in[8];
  const float* b1     = (const float*)d_in[9];
  const float* W2     = (const float*)d_in[10];
  const float* b2     = (const float*)d_in[11];
  float* out = (float*)d_out;

  char* ws = (char*)d_ws;
  const size_t fzB  = (size_t)RTOT * MD * sizeof(float);      // 64 MB
  const size_t x16B = (size_t)RTOT * MD * sizeof(_Float16);   // 32 MB
  const size_t h16B = x16B;                                   // 32 MB
  const size_t stB  = (size_t)BB * CCH * MD * sizeof(float2); // 2 MB
  float*     fz  = (float*)ws;
  _Float16*  x16 = (_Float16*)(ws + fzB);
  _Float16*  h16 = (_Float16*)(ws + fzB + x16B);
  float2*    S   = (float2*)(ws + fzB + x16B + h16B);
  float2*    Vin = (float2*)(ws + fzB + x16B + h16B + stB);

  const int gemmBlocks = RTOT / 128;        // 1024
  const int scanBlocks = (BB * CCH * MD) / 256; // 1024
  const size_t lds1 = (size_t)MD * MD * sizeof(_Float16);     // 32 KB
  const size_t lds2 = 2 * lds1;                               // 64 KB

  k_gemm_fz  <<<gemmBlocks, 256, lds1, stream>>>(u, B_w, B_b, fz);
  k_scan_local<<<scanBlocks, 256, 0,   stream>>>(fz, A_diag, S);
  k_scan_prefix<<<(BB * MD) / 256, 256, 0, stream>>>(A_diag, S, Vin);
  k_scan_replay<<<scanBlocks, 256, 0,  stream>>>(fz, A_diag, Vin, x16);
  k_gemm_h   <<<gemmBlocks, 256, lds2, stream>>>(x16, u, C_w, C_b, D_w, D_b, h16);
  k_gemm_out <<<gemmBlocks, 256, lds2, stream>>>(h16, u, W1, b1, W2, b2, out);
}
